// MultiheadSelfAttention_13692355739715
// MI455X (gfx1250) — matrix operand order repaired and checked by reading, not compiled, not run on hardware
//
#include <hip/hip_runtime.h>
#include <hip/hip_bf16.h>

#define D_MODEL   1024
#define NUM_HEADS 16
#define D_K       64
#define SEQ       2048
#define BATCH     4
#define M_TOTAL   (BATCH * SEQ)   // 8192 rows

typedef __bf16 bf16_t;
typedef __attribute__((ext_vector_type(8)))  __bf16 v8bf;
typedef __attribute__((ext_vector_type(16))) __bf16 v16bf;
typedef __attribute__((ext_vector_type(8)))  float  v8f;
typedef __attribute__((ext_vector_type(4)))  float  v4f;
typedef __attribute__((ext_vector_type(4)))  unsigned int v4u;
typedef __attribute__((ext_vector_type(8)))  int v8i;
typedef __attribute__((ext_vector_type(4)))  int v4i;

__device__ inline v8f v8f_zero() {
  v8f z = {0.f,0.f,0.f,0.f,0.f,0.f,0.f,0.f};
  return z;
}

// A fragment (16x32 bf16): lanes 0-15 row M=lane, chunks {k0..k0+7, k0+16..k0+23}, k0 = 8*(lane>=16)
__device__ inline v16bf load_frag_a(const bf16_t* lds, int stride, int kbase, int lane) {
  int m  = lane & 15;
  int k0 = (lane >> 4) ? 8 : 0;
  const bf16_t* p = lds + m * stride + kbase + k0;
  union { v16bf v; v8bf h[2]; } u;
  u.h[0] = *(const v8bf*)p;
  u.h[1] = *(const v8bf*)(p + 16);
  return u.v;
}

// B fragment (32x16 bf16): col N = nbase+(lane&15), chunks {k0..k0+7, k0+16..k0+23}, k0 = 8*(lane>=16).
// LDS stored [n][k] (contiguous in k).
__device__ inline v16bf load_frag_b(const bf16_t* lds, int stride, int nbase, int kbase, int lane) {
  int n  = nbase + (lane & 15);
  int k0 = (lane >> 4) ? 8 : 0;
  const bf16_t* p = lds + n * stride + kbase + k0;
  union { v16bf v; v8bf h[2]; } u;
  u.h[0] = *(const v8bf*)p;
  u.h[1] = *(const v8bf*)(p + 16);
  return u.v;
}

#define WMMA_BF16(a, b, c) \
  __builtin_amdgcn_wmma_f32_16x16x32_bf16(false, (a), false, (b), (short)0, (c), false, false)

// ---------------------------------------------------------------------------
// Tensor Data Mover: 2-D bf16 tile (tile_d1 rows x tile_d0 contiguous elems)
// from global (row stride d0_stride elems) into contiguous LDS.
// Descriptor bit layout per CDNA5 ISA sec 8.3/8.4 (D# group0/group1).
// 6-arg builtin (clang-23 lane): groups 2/3 (v4i) + extra v8i group, then cpol.
// ---------------------------------------------------------------------------
__device__ inline void tdm_load_2d(unsigned lds_addr, unsigned long long gaddr,
                                   unsigned tensor_d0, unsigned tensor_d1,
                                   unsigned tile_d0, unsigned tile_d1,
                                   unsigned long long d0_stride)
{
  v4u g0;
  g0[0] = 1u;                                            // count=1, user mode
  g0[1] = lds_addr;                                      // [63:32] lds byte addr
  g0[2] = (unsigned)(gaddr & 0xffffffffu);               // [95:64] global addr lo
  g0[3] = (unsigned)((gaddr >> 32) & 0x1ffffffu)         // [120:96] global addr hi
        | (2u << 30);                                    // [127:126] type=2 (image)
  v8i g1;
  g1[0] = (int)(1u << 16);                               // data_size=1 -> 2 bytes/elem
  g1[1] = (int)((tensor_d0 & 0xffffu) << 16);            // tensor_dim0[15:0] @ bit48
  g1[2] = (int)(((tensor_d0 >> 16) & 0xffffu)
        | ((tensor_d1 & 0xffffu) << 16));                // dim0 hi | dim1 lo
  g1[3] = (int)(((tensor_d1 >> 16) & 0xffffu)
        | ((tile_d0 & 0xffffu) << 16));                  // dim1 hi | tile_dim0
  g1[4] = (int)(tile_d1 & 0xffffu);                      // tile_dim1 (tile_dim2=0)
  g1[5] = (int)(unsigned)(d0_stride & 0xffffffffu);      // tensor_dim0_stride lo
  g1[6] = (int)(unsigned)((d0_stride >> 32) & 0xffffu);  // stride hi (dim1_stride=0)
  g1[7] = 0;
  v4i z4 = {0, 0, 0, 0};                                 // groups 2/3 unused (2-D)
  v8i z8 = {0, 0, 0, 0, 0, 0, 0, 0};
  __builtin_amdgcn_tensor_load_to_lds(g0, g1, z4, z4, z8, 0);
}

// ---------------------------------------------------------------------------
// Kernel 1: Q/K/V = X * W^T (bf16 WMMA, fp32 accum), fused RoPE.
// Q,K -> bf16 [B,H,S,64];  V -> bf16 [B,H,64,S] (pre-transposed for P.V tiles)
// grid: (NUM_HEADS, M_TOTAL/64, 3), block 128
// ---------------------------------------------------------------------------
__global__ __launch_bounds__(128)
void qkv_rope_gemm(const float* __restrict__ X,  const float* __restrict__ Wq,
                   const float* __restrict__ Wk, const float* __restrict__ Wv,
                   const int* __restrict__ tok,
                   bf16_t* __restrict__ Qb, bf16_t* __restrict__ Kb, bf16_t* __restrict__ Vt)
{
  __shared__ __align__(16) bf16_t ldsA[64 * 40];
  __shared__ __align__(16) bf16_t ldsB[64 * 40];
  __shared__ __align__(16) float  stg [64 * 68];

  const int tid   = threadIdx.x;
  const int lane  = tid & 31;
  const int wave  = tid >> 5;
  const int h     = blockIdx.x;           // head == 64-wide N tile
  const int m0    = blockIdx.y * 64;
  const int which = blockIdx.z;           // 0=Q 1=K 2=V
  const float* W  = (which == 0) ? Wq : ((which == 1) ? Wk : Wv);
  const int n0 = h * 64;

  v8f acc[4];
  #pragma unroll
  for (int nt = 0; nt < 4; ++nt) acc[nt] = v8f_zero();

  const int row = tid >> 1;            // 0..63
  const int cc  = (tid & 1) * 16;      // 0 / 16

  for (int kt = 0; kt < D_MODEL / 32; ++kt) {
    const float* sa = X + (size_t)(m0 + row) * D_MODEL + kt * 32 + cc;
    const float* sb = W + (size_t)(n0 + row) * D_MODEL + kt * 32 + cc;
    if (kt + 1 < D_MODEL / 32) {       // global_prefetch_b8 of next K tile
      __builtin_prefetch(sa + 32, 0, 0);
      __builtin_prefetch(sb + 32, 0, 0);
    }
    #pragma unroll
    for (int j = 0; j < 4; ++j) {
      v4f ta = *(const v4f*)(sa + 4 * j);
      v4f tb = *(const v4f*)(sb + 4 * j);
      int o = row * 40 + cc + 4 * j;
      ldsA[o + 0] = (bf16_t)ta.x; ldsA[o + 1] = (bf16_t)ta.y;
      ldsA[o + 2] = (bf16_t)ta.z; ldsA[o + 3] = (bf16_t)ta.w;
      ldsB[o + 0] = (bf16_t)tb.x; ldsB[o + 1] = (bf16_t)tb.y;
      ldsB[o + 2] = (bf16_t)tb.z; ldsB[o + 3] = (bf16_t)tb.w;
    }
    __syncthreads();

    v16bf a = load_frag_a(ldsA + wave * 16 * 40, 40, 0, lane);
    #pragma unroll
    for (int nt = 0; nt < 4; ++nt) {
      v16bf b = load_frag_b(ldsB, 40, nt * 16, 0, lane);
      acc[nt] = WMMA_BF16(a, b, acc[nt]);
    }
    __syncthreads();
  }

  // stage fp32 tile (D layout: row = vgpr + 8*(lane>=16), col = lane&15)
  {
    const int half8 = (lane >> 4) * 8;
    const int n     = lane & 15;
    #pragma unroll
    for (int nt = 0; nt < 4; ++nt)
      #pragma unroll
      for (int r = 0; r < 8; ++r)
        stg[(wave * 16 + half8 + r) * 68 + nt * 16 + n] = acc[nt][r];
  }
  __syncthreads();

  if (which < 2) {
    bf16_t* dst = (which == 0) ? Qb : Kb;
    for (int it = tid; it < 64 * 32; it += 128) {
      int mr = it >> 5, pi = it & 31;
      int g = m0 + mr;
      int bb = g >> 11;                   // / SEQ
      int s  = g & (SEQ - 1);
      float pos  = (float)tok[s];
      float freq = __powf(10000.0f, -(float)pi / 32.0f);  // theta^(-2*pi/64)
      float sn, cs;
      __sincosf(pos * freq, &sn, &cs);
      float e = stg[mr * 68 + 2 * pi];
      float o = stg[mr * 68 + 2 * pi + 1];
      size_t base = (((size_t)bb * NUM_HEADS + h) * SEQ + s) * D_K + 2 * pi;
      dst[base]     = (bf16_t)(e * cs - o * sn);
      dst[base + 1] = (bf16_t)(e * sn + o * cs);
    }
  } else {
    // V transposed per head: Vt[((b*H + h)*64 + d)*SEQ + s]
    for (int it = tid; it < 64 * 64; it += 128) {
      int mr = it >> 6, d = it & 63;
      int g = m0 + mr;
      int bb = g >> 11, s = g & (SEQ - 1);
      Vt[(((size_t)bb * NUM_HEADS + h) * D_K + d) * SEQ + s] = (bf16_t)stg[mr * 68 + d];
    }
  }
}

// ---------------------------------------------------------------------------
// Kernel 2: causal flash attention; K/V tiles via Tensor Data Mover.
// grid: (SEQ/64, NUM_HEADS, BATCH), block 128 (4 waves x 16 query rows)
// ---------------------------------------------------------------------------
__global__ __launch_bounds__(128)
void flash_attn(const bf16_t* __restrict__ Qb, const bf16_t* __restrict__ Kb,
                const bf16_t* __restrict__ Vt, bf16_t* __restrict__ AO)
{
  __shared__ __align__(16) bf16_t ldsK [64 * 64];      // [key][d]   (TDM tile)
  __shared__ __align__(16) bf16_t ldsV [64 * 64];      // [d][key]   (TDM tile)
  __shared__ __align__(16) bf16_t ldsP [4 * 16 * 72];  // per-wave P staging

  const int tid  = threadIdx.x, lane = tid & 31, wave = tid >> 5;
  const int qt = blockIdx.x, h = blockIdx.y, b = blockIdx.z;
  const int q0 = qt * 64 + wave * 16;
  const size_t headbase = ((size_t)b * NUM_HEADS + h) * SEQ;       // rows of Q/K
  const size_t vhead    = ((size_t)b * NUM_HEADS + h) * D_K * SEQ; // Vt head base
  const int half8 = (lane >> 4) * 8;
  const float NEG = -3.0e38f;

  // preload Q A-fragments (two K-steps over d)
  v16bf qA[2];
  {
    int m  = lane & 15;
    int k0 = (lane >> 4) ? 8 : 0;
    const bf16_t* qp = Qb + (headbase + q0 + m) * D_K;
    #pragma unroll
    for (int sp = 0; sp < 2; ++sp) {
      union { v16bf v; v8bf hh[2]; } u;
      u.hh[0] = *(const v8bf*)(qp + sp * 32 + k0);
      u.hh[1] = *(const v8bf*)(qp + sp * 32 + k0 + 16);
      qA[sp] = u.v;
    }
  }

  v8f Oacc[4];
  #pragma unroll
  for (int nt = 0; nt < 4; ++nt) Oacc[nt] = v8f_zero();
  v8f m_i = {NEG,NEG,NEG,NEG,NEG,NEG,NEG,NEG};
  v8f l_i = v8f_zero();

  bf16_t* pbuf = ldsP + wave * 16 * 72;
  const unsigned la_k = (unsigned)(uintptr_t)&ldsK[0];
  const unsigned la_v = (unsigned)(uintptr_t)&ldsV[0];

  for (int kt = 0; kt <= qt; ++kt) {
    const int ktb = kt * 64;

    // TDM: whole-tile DMA, issued once (EXEC-independent), tracked by TENSORcnt
    if (wave == 0) {
      tdm_load_2d(la_k,
                  (unsigned long long)(uintptr_t)(Kb + (headbase + ktb) * D_K),
                  64u, (unsigned)SEQ, 64u, 64u, 64ull);
      tdm_load_2d(la_v,
                  (unsigned long long)(uintptr_t)(Vt + vhead + ktb),
                  (unsigned)SEQ, 64u, 64u, 64u, (unsigned long long)SEQ);
      __builtin_amdgcn_s_wait_tensorcnt(0);
    }
    __syncthreads();

    // S = Q.K^T  (16x64 per wave)
    v8f s[4];
    #pragma unroll
    for (int nt = 0; nt < 4; ++nt) s[nt] = v8f_zero();
    #pragma unroll
    for (int sp = 0; sp < 2; ++sp)
      #pragma unroll
      for (int nt = 0; nt < 4; ++nt) {
        v16bf bfr = load_frag_b(ldsK, 64, nt * 16, sp * 32, lane);
        s[nt] = WMMA_BF16(qA[sp], bfr, s[nt]);
      }

    // scale + causal mask
    #pragma unroll
    for (int nt = 0; nt < 4; ++nt) {
      int kg = ktb + nt * 16 + (lane & 15);
      #pragma unroll
      for (int r = 0; r < 8; ++r) {
        float v = s[nt][r] * 0.125f;                 // 1/sqrt(64)
        s[nt][r] = (kg > q0 + half8 + r) ? NEG : v;
      }
    }

    // online softmax: row max (xor-reduce inside 16-lane half)
    v8f mnew = m_i;
    #pragma unroll
    for (int nt = 0; nt < 4; ++nt)
      #pragma unroll
      for (int r = 0; r < 8; ++r) mnew[r] = fmaxf(mnew[r], s[nt][r]);
    #pragma unroll
    for (int off = 1; off <= 8; off <<= 1)
      #pragma unroll
      for (int r = 0; r < 8; ++r)
        mnew[r] = fmaxf(mnew[r], __shfl_xor(mnew[r], off, 32));

    // P = exp(S - m): row sums + bf16 staging in A-matrix layout
    v8f lnew = v8f_zero();
    #pragma unroll
    for (int nt = 0; nt < 4; ++nt)
      #pragma unroll
      for (int r = 0; r < 8; ++r) {
        float p = __expf(s[nt][r] - mnew[r]);
        lnew[r] += p;
        pbuf[(half8 + r) * 72 + nt * 16 + (lane & 15)] = (bf16_t)p;
      }
    #pragma unroll
    for (int off = 1; off <= 8; off <<= 1)
      #pragma unroll
      for (int r = 0; r < 8; ++r) lnew[r] += __shfl_xor(lnew[r], off, 32);

    // rescale running state
    #pragma unroll
    for (int r = 0; r < 8; ++r) {
      float f = __expf(m_i[r] - mnew[r]);
      l_i[r] = l_i[r] * f + lnew[r];
      #pragma unroll
      for (int nt = 0; nt < 4; ++nt) Oacc[nt][r] *= f;
    }
    m_i = mnew;
    __syncthreads();

    // O += P.V
    #pragma unroll
    for (int sp = 0; sp < 2; ++sp) {
      v16bf pa = load_frag_a(pbuf, 72, sp * 32, lane);
      #pragma unroll
      for (int nt = 0; nt < 4; ++nt) {
        v16bf vb = load_frag_b(ldsV, 64, nt * 16, sp * 32, lane);
        Oacc[nt] = WMMA_BF16(pa, vb, Oacc[nt]);
      }
    }
    __syncthreads();
  }

  // epilogue: O / l, write bf16 [B,S, h*64+d] (== d_model layout for final GEMM)
  #pragma unroll
  for (int nt = 0; nt < 4; ++nt)
    #pragma unroll
    for (int r = 0; r < 8; ++r) {
      int q = q0 + half8 + r;
      float o = Oacc[nt][r] / l_i[r];
      AO[((size_t)(b * SEQ + q)) * D_MODEL + h * D_K + nt * 16 + (lane & 15)] = (bf16_t)o;
    }
}

// ---------------------------------------------------------------------------
// Kernel 3: out = AO * W_O^T; A tile via async-to-LDS (ASYNCcnt path)
// grid: (NUM_HEADS, M_TOTAL/64), block 128
// ---------------------------------------------------------------------------
__global__ __launch_bounds__(128)
void out_proj(const bf16_t* __restrict__ A, const float* __restrict__ Wo,
              float* __restrict__ out)
{
  __shared__ __align__(16) bf16_t ldsA[64 * 40];
  __shared__ __align__(16) bf16_t ldsB[64 * 40];

  const int tid = threadIdx.x, lane = tid & 31, wave = tid >> 5;
  const int n0 = blockIdx.x * 64;
  const int m0 = blockIdx.y * 64;

  v8f acc[4];
  #pragma unroll
  for (int nt = 0; nt < 4; ++nt) acc[nt] = v8f_zero();

  const int row = tid >> 1;
  const int cc  = (tid & 1) * 16;
  const unsigned la = (unsigned)(uintptr_t)&ldsA[row * 40 + cc];

  for (int kt = 0; kt < D_MODEL / 32; ++kt) {
    const bf16_t* sa = A  + (size_t)(m0 + row) * D_MODEL + kt * 32 + cc;
    const float*  sb = Wo + (size_t)(n0 + row) * D_MODEL + kt * 32 + cc;
    if (kt + 1 < D_MODEL / 32) {
      __builtin_prefetch(sa + 32, 0, 0);
      __builtin_prefetch(sb + 32, 0, 0);
    }
    // A tile: per-lane async copy global->LDS (32 bytes), no VGPR staging
    {
      unsigned long long ga = (unsigned long long)(uintptr_t)sa;
      asm volatile("global_load_async_to_lds_b128 %0, %1, off"
                   :: "v"(la), "v"(ga) : "memory");
      asm volatile("global_load_async_to_lds_b128 %0, %1, off offset:16"
                   :: "v"(la), "v"(ga) : "memory");
    }
    // B tile: fp32 -> bf16 conversion path
    #pragma unroll
    for (int j = 0; j < 4; ++j) {
      v4f tb = *(const v4f*)(sb + 4 * j);
      int o = row * 40 + cc + 4 * j;
      ldsB[o + 0] = (bf16_t)tb.x; ldsB[o + 1] = (bf16_t)tb.y;
      ldsB[o + 2] = (bf16_t)tb.z; ldsB[o + 3] = (bf16_t)tb.w;
    }
    asm volatile("s_wait_asynccnt 0" ::: "memory");
    __syncthreads();

    v16bf a = load_frag_a(ldsA + wave * 16 * 40, 40, 0, lane);
    #pragma unroll
    for (int nt = 0; nt < 4; ++nt) {
      v16bf b = load_frag_b(ldsB, 40, nt * 16, 0, lane);
      acc[nt] = WMMA_BF16(a, b, acc[nt]);
    }
    __syncthreads();
  }

  const int half8 = (lane >> 4) * 8;
  const int n     = lane & 15;
  #pragma unroll
  for (int nt = 0; nt < 4; ++nt)
    #pragma unroll
    for (int r = 0; r < 8; ++r)
      out[(size_t)(m0 + wave * 16 + half8 + r) * D_MODEL + n0 + nt * 16 + n] = acc[nt][r];
}

// ---------------------------------------------------------------------------
extern "C" void kernel_launch(void* const* d_in, const int* in_sizes, int n_in,
                              void* d_out, int out_size, void* d_ws, size_t ws_size,
                              hipStream_t stream)
{
  const float* x   = (const float*)d_in[0];
  const float* Wq  = (const float*)d_in[1];
  const float* Wk  = (const float*)d_in[2];
  const float* Wv  = (const float*)d_in[3];
  const float* Wo  = (const float*)d_in[4];
  const int*   tok = (const int*)  d_in[5];
  float* out = (float*)d_out;

  const size_t HBUF = (size_t)BATCH * NUM_HEADS * SEQ * D_K;  // 8,388,608 elems
  bf16_t* Qb  = (bf16_t*)d_ws;
  bf16_t* Kb  = Qb  + HBUF;
  bf16_t* Vtb = Kb  + HBUF;   // per-head transposed V
  bf16_t* AO  = Vtb + HBUF;

  qkv_rope_gemm<<<dim3(NUM_HEADS, M_TOTAL / 64, 3), 128, 0, stream>>>(
      x, Wq, Wk, Wv, tok, Qb, Kb, Vtb);
  flash_attn<<<dim3(SEQ / 64, NUM_HEADS, BATCH), 128, 0, stream>>>(
      Qb, Kb, Vtb, AO);
  out_proj<<<dim3(NUM_HEADS, M_TOTAL / 64), 128, 0, stream>>>(
      AO, Wo, out);
}
